// DLI_loss_3_6614249636378
// MI455X (gfx1250) — matrix operand order repaired and checked by reading, not compiled, not run on hardware
//
#include <hip/hip_runtime.h>
#include <hip/hip_bf16.h>
#include <math.h>

// ---------------------------------------------------------------------------
// Problem constants (from reference)
// ---------------------------------------------------------------------------
#define ENC_DIM   1024
#define LSTM_HID  1024
#define BSZ       32
#define TT        64
#define JJ        61            // T - 3
#define NSEQ      (BSZ * JJ)    // 1952
#define NSTATE    (BSZ * TT)    // 2048
#define GATES     (4 * LSTM_HID) // 4096
#define MPAD      2048          // padded M for GEMMs (>= NSEQ, multiple of 32)

typedef __attribute__((ext_vector_type(16))) __bf16 v16bf;
typedef __attribute__((ext_vector_type(8)))  __bf16 v8bf;
typedef __attribute__((ext_vector_type(8)))  float  v8f;

__device__ __forceinline__ __bf16 f2bf(float f) {
    unsigned u = __builtin_bit_cast(unsigned, f);
    unsigned r = u + 0x7FFFu + ((u >> 16) & 1u);   // round-to-nearest-even
    unsigned short h = (unsigned short)(r >> 16);
    return __builtin_bit_cast(__bf16, h);
}

__device__ __forceinline__ float sigm(float x) { return 1.0f / (1.0f + expf(-x)); }

// ---------------------------------------------------------------------------
// Kernel 0: convert LSTM weights f32 -> bf16, fuse biases
// ---------------------------------------------------------------------------
__global__ void cvt_weights(const float* __restrict__ Wih, const float* __restrict__ Whh,
                            __bf16* __restrict__ WihB, __bf16* __restrict__ WhhB,
                            const float* __restrict__ bih, const float* __restrict__ bhh,
                            float* __restrict__ bias) {
    size_t gid = (size_t)blockIdx.x * blockDim.x + threadIdx.x;   // GATES*ENC_DIM total
    WihB[gid] = f2bf(Wih[gid]);
    WhhB[gid] = f2bf(Whh[gid]);
    if (gid < GATES) bias[gid] = bih[gid] + bhh[gid];
}

// ---------------------------------------------------------------------------
// Kernel 1: gather states = encoder_output[b, ids[b,t], :]  (f32 + bf16 copy)
// ---------------------------------------------------------------------------
__global__ void gather_states(const float* __restrict__ enc, const int* __restrict__ ids,
                              float* __restrict__ stF, __bf16* __restrict__ stB) {
    size_t gid = (size_t)blockIdx.x * blockDim.x + threadIdx.x;   // NSTATE*ENC_DIM total
    int r = (int)(gid >> 10);          // state row: b*64 + t
    int e = (int)(gid & 1023);
    int b = r >> 6;
    int id = ids[r];
    float v = enc[((size_t)b * 1024 + (size_t)id) * ENC_DIM + e];
    stF[gid] = v;
    stB[gid] = f2bf(v);
}

// ---------------------------------------------------------------------------
// Kernel 2: WMMA GEMM  C[M,N] = A[M,K] * B[N,K]^T (+ per-column bias)
//   A: bf16 row-major (lda = K), B: bf16 row-major (N rows of K) -> B^T fragment
//   One wave -> 32 (M) x 64 (N) stripe: 2 A-fragments x 4 B-tiles = 8 WMMA
//   accumulators; each B fragment is reused by 2 WMMAs (halves B-side L2
//   traffic per FLOP vs a 16x64 tile). K-loop step 32.
//   A/B fragments use contiguous k = 16*(lane/16) + s in both; C/D per ISA (wave32).
// ---------------------------------------------------------------------------
__global__ __launch_bounds__(128)
void wmma_gemm_bf16(const __bf16* __restrict__ A, int lda,
                    const __bf16* __restrict__ B, int ldb,
                    const float* __restrict__ bias,
                    float* __restrict__ C, int ldc,
                    int nWaveCols, int K) {
    const int lane = threadIdx.x & 31;
    const int wave = blockIdx.x * (blockDim.x >> 5) + (threadIdx.x >> 5);
    const int wm = wave / nWaveCols;
    const int wn = wave - wm * nWaveCols;
    const size_t m0 = (size_t)wm * 32;
    const size_t n0 = (size_t)wn * 64;

    v8f acc[8];
    #pragma unroll
    for (int i = 0; i < 8; ++i) acc[i] = (v8f){};

    // A fragment addressing: lane (0..15)->row m, lanes 16..31 carry K+16 half
    const int mA    = lane & 15;
    const int koffA = (lane >> 4) * 16;
    const __bf16* Arow0 = A + (m0 + mA) * (size_t)lda;         // rows m0..m0+15
    const __bf16* Arow1 = Arow0 + (size_t)16 * lda;            // rows m0+16..m0+31

    // B fragment addressing: lane (0..15)->col n, lanes 16..31 carry K+16 half;
    // contiguous K within a row of B (since GEMM is A * B^T with B row-major)
    const int nB    = lane & 15;
    const int koffB = (lane >> 4) << 4;
    const __bf16* Brow0 = B + (n0 + nB) * (size_t)ldb;
    const __bf16* Brow1 = Brow0 + (size_t)16 * ldb;
    const __bf16* Brow2 = Brow0 + (size_t)32 * ldb;
    const __bf16* Brow3 = Brow0 + (size_t)48 * ldb;

    #pragma unroll 2
    for (int k0 = 0; k0 < K; k0 += 32) {
        // prefetch next k-step of A into cache hierarchy
        __builtin_prefetch(Arow0 + k0 + 32, 0, 3);
        __builtin_prefetch(Arow1 + k0 + 32, 0, 3);

        v8bf alo0 = *(const v8bf*)(Arow0 + k0 + koffA);        // K = k0+koff .. +7
        v8bf ahi0 = *(const v8bf*)(Arow0 + k0 + 8 + koffA);    // K = k0+8+koff .. +7
        v8bf alo1 = *(const v8bf*)(Arow1 + k0 + koffA);
        v8bf ahi1 = *(const v8bf*)(Arow1 + k0 + 8 + koffA);
        v16bf afrag0, afrag1;
        #pragma unroll
        for (int e = 0; e < 8; ++e) {
            afrag0[e] = alo0[e]; afrag0[8 + e] = ahi0[e];
            afrag1[e] = alo1[e]; afrag1[8 + e] = ahi1[e];
        }

        v16bf bfrag[4];
        bfrag[0] = *(const v16bf*)(Brow0 + k0 + koffB);
        bfrag[1] = *(const v16bf*)(Brow1 + k0 + koffB);
        bfrag[2] = *(const v16bf*)(Brow2 + k0 + koffB);
        bfrag[3] = *(const v16bf*)(Brow3 + k0 + koffB);

        #pragma unroll
        for (int t = 0; t < 4; ++t) {
            acc[t]     = __builtin_amdgcn_wmma_f32_16x16x32_bf16(false, afrag0, false, bfrag[t],
                                                                 (short)0, acc[t], false, false);
            acc[4 + t] = __builtin_amdgcn_wmma_f32_16x16x32_bf16(false, afrag1, false, bfrag[t],
                                                                 (short)0, acc[4 + t], false, false);
        }
    }

    // C/D layout: VGPR r -> row mBase + (lane>>4)*8 + r, col n0 + t*16 + (lane&15)
    const int nn = lane & 15;
    const int mm = (lane >> 4) * 8;
    float bv[4];
    #pragma unroll
    for (int t = 0; t < 4; ++t) bv[t] = bias ? bias[n0 + t * 16 + nn] : 0.0f;
    #pragma unroll
    for (int half = 0; half < 2; ++half) {
        #pragma unroll
        for (int r = 0; r < 8; ++r) {
            size_t row = (m0 + half * 16 + mm + r) * (size_t)ldc + n0 + nn;
            #pragma unroll
            for (int t = 0; t < 4; ++t)
                C[row + t * 16] = acc[half * 4 + t][r] + bv[t];
        }
    }
}

// ---------------------------------------------------------------------------
// Kernel 3: LSTM cell (elementwise). t=0 uses h0=c0=0 (no recurrent term).
// ---------------------------------------------------------------------------
__global__ void lstm_cell(const float* __restrict__ Xproj, const float* __restrict__ Gh,
                          float* c_state, float* __restrict__ h_out,
                          __bf16* __restrict__ h_bf, int t) {
    int gid = blockIdx.x * blockDim.x + threadIdx.x;
    if (gid >= NSEQ * LSTM_HID) return;
    int s = gid >> 10;
    int u = gid & 1023;
    int b = s / JJ;
    int j = s - b * JJ;
    const float* g = Xproj + ((size_t)(b * TT + j + t)) * GATES;
    float gi = g[u], gf = g[LSTM_HID + u], gg = g[2 * LSTM_HID + u], go = g[3 * LSTM_HID + u];
    float cp = 0.0f;
    if (t > 0) {
        const float* gh = Gh + (size_t)s * GATES;
        gi += gh[u]; gf += gh[LSTM_HID + u]; gg += gh[2 * LSTM_HID + u]; go += gh[3 * LSTM_HID + u];
        cp = c_state[gid];
    }
    float iv = sigm(gi), fv = sigm(gf), gv = tanhf(gg), ov = sigm(go);
    float c = fv * cp + iv * gv;
    float h = ov * tanhf(c);
    c_state[gid] = c;
    h_out[gid] = h;
    h_bf[gid] = f2bf(h);
}

// ---------------------------------------------------------------------------
// Kernel 4: per-row dot product  out[r] = X[r,:] . w   (block per row)
// ---------------------------------------------------------------------------
__global__ void row_dot(const float* __restrict__ X, int ldx,
                        const float* __restrict__ w, int n, float* __restrict__ out) {
    const float* x = X + (size_t)blockIdx.x * ldx;
    float s = 0.0f;
    for (int u = threadIdx.x; u < n; u += blockDim.x) s += x[u] * w[u];
    __shared__ float sd[256];
    sd[threadIdx.x] = s;
    __syncthreads();
    for (int off = 128; off > 0; off >>= 1) {
        if ((int)threadIdx.x < off) sd[threadIdx.x] += sd[threadIdx.x + off];
        __syncthreads();
    }
    if (threadIdx.x == 0) out[blockIdx.x] = sd[0];
}

// ---------------------------------------------------------------------------
// Kernel 5: masked logsumexp loss per (b,j). Block = 64 threads over m.
// ---------------------------------------------------------------------------
__global__ void loss_kernel(const float* __restrict__ a, const float* __restrict__ c_all,
                            const float* __restrict__ fcb, float* __restrict__ loss) {
    int s = blockIdx.x;           // 0..NSEQ-1
    int b = s / JJ;
    int j = s - b * JJ;
    int m = threadIdx.x;          // 0..63
    float av = a[s];
    float fb = fcb[0];
    const float NEG = -1e30f;
    float val = NEG;
    if (m < JJ) {
        bool valid = (j + m) <= (TT - 4);
        int tgt = j + 3 + m; if (tgt > TT - 1) tgt = TT - 1;
        float lg = av + c_all[b * TT + tgt] + fb;
        val = valid ? lg : NEG;
    }
    __shared__ float sd[64];
    sd[m] = val;
    __syncthreads();
    for (int off = 32; off > 0; off >>= 1) {
        if (m < off) sd[m] = fmaxf(sd[m], sd[m + off]);
        __syncthreads();
    }
    float mx = sd[0];
    __syncthreads();
    sd[m] = expf(val - mx);
    __syncthreads();
    for (int off = 32; off > 0; off >>= 1) {
        if (m < off) sd[m] += sd[m + off];
        __syncthreads();
    }
    if (m == 0) {
        float lse = mx + logf(sd[0]);
        float logit0 = av + c_all[b * TT + j + 3] + fb;
        loss[s] = lse - logit0;
    }
}

// ---------------------------------------------------------------------------
// Kernel 6: mean over NSEQ losses -> d_out[0]
// ---------------------------------------------------------------------------
__global__ void reduce_mean(const float* __restrict__ loss, float* __restrict__ out) {
    float s = 0.0f;
    for (int i = threadIdx.x; i < NSEQ; i += 256) s += loss[i];
    __shared__ float sd[256];
    sd[threadIdx.x] = s;
    __syncthreads();
    for (int off = 128; off > 0; off >>= 1) {
        if ((int)threadIdx.x < off) sd[threadIdx.x] += sd[threadIdx.x + off];
        __syncthreads();
    }
    if (threadIdx.x == 0) out[0] = sd[0] / (float)NSEQ;
}

// ---------------------------------------------------------------------------
// Workspace layout (bytes)
// ---------------------------------------------------------------------------
#define OFF_STF   ((size_t)0)                          // states f32:   2048*1024*4
#define OFF_STB   (OFF_STF  + (size_t)8388608)         // states bf16:  2048*1024*2
#define OFF_WIHB  (OFF_STB  + (size_t)4194304)         // W_ih bf16:    4096*1024*2
#define OFF_WHHB  (OFF_WIHB + (size_t)8388608)         // W_hh bf16:    4096*1024*2
#define OFF_BIAS  (OFF_WHHB + (size_t)8388608)         // bias:         4096*4
#define OFF_XPROJ (OFF_BIAS + (size_t)16384)           // Xproj:        2048*4096*4
#define OFF_GH    (OFF_XPROJ+ (size_t)33554432)        // Gh:           2048*4096*4
#define OFF_H     (OFF_GH   + (size_t)33554432)        // h f32:        2048*1024*4
#define OFF_CST   (OFF_H    + (size_t)8388608)         // c f32:        2048*1024*4
#define OFF_HBF   (OFF_CST  + (size_t)8388608)         // h bf16:       2048*1024*2
#define OFF_A     (OFF_HBF  + (size_t)4194304)         // a:            2048*4
#define OFF_CALL  (OFF_A    + (size_t)8192)            // c_all:        2048*4
#define OFF_LOSS  (OFF_CALL + (size_t)8192)            // loss:         2048*4

extern "C" void kernel_launch(void* const* d_in, const int* in_sizes, int n_in,
                              void* d_out, int out_size, void* d_ws, size_t ws_size,
                              hipStream_t stream) {
    (void)in_sizes; (void)n_in; (void)out_size; (void)ws_size;
    const float* enc  = (const float*)d_in[0];
    const int*   ids  = (const int*)  d_in[1];
    const float* W_ih = (const float*)d_in[2];
    const float* W_hh = (const float*)d_in[3];
    const float* b_ih = (const float*)d_in[4];
    const float* b_hh = (const float*)d_in[5];
    const float* fc_w = (const float*)d_in[6];   // [0:1024]=Wh, [1024:2048]=We
    const float* fc_b = (const float*)d_in[7];
    float* out = (float*)d_out;
    char*  ws  = (char*)d_ws;

    float*  stF   = (float*) (ws + OFF_STF);
    __bf16* stB   = (__bf16*)(ws + OFF_STB);
    __bf16* WihB  = (__bf16*)(ws + OFF_WIHB);
    __bf16* WhhB  = (__bf16*)(ws + OFF_WHHB);
    float*  bias  = (float*) (ws + OFF_BIAS);
    float*  Xproj = (float*) (ws + OFF_XPROJ);
    float*  Gh    = (float*) (ws + OFF_GH);
    float*  hF    = (float*) (ws + OFF_H);
    float*  cSt   = (float*) (ws + OFF_CST);
    __bf16* hB    = (__bf16*)(ws + OFF_HBF);
    float*  aV    = (float*) (ws + OFF_A);
    float*  cAll  = (float*) (ws + OFF_CALL);
    float*  loss  = (float*) (ws + OFF_LOSS);

    // 0) weight conversion + bias fuse  (4096*1024 threads)
    cvt_weights<<<(GATES * ENC_DIM) / 256, 256, 0, stream>>>(W_ih, W_hh, WihB, WhhB, b_ih, b_hh, bias);

    // 1) gather states (2048*1024 threads)
    gather_states<<<(NSTATE * ENC_DIM) / 256, 256, 0, stream>>>(enc, ids, stF, stB);

    // GEMM geometry: M=2048, N=4096, K=1024
    //   -> 64 wave-rows (32 M each) x 64 wave-cols (64 N each), 4 waves/block
    const int waveCols = GATES / 64;                          // 64
    const int gemmBlocks = (MPAD / 32) * waveCols / 4;        // 1024

    // 2) Xproj = states @ W_ih^T + (b_ih + b_hh)
    wmma_gemm_bf16<<<gemmBlocks, 128, 0, stream>>>(stB, ENC_DIM, WihB, ENC_DIM, bias,
                                                   Xproj, GATES, waveCols, ENC_DIM);

    const int cellBlocks = (NSEQ * LSTM_HID + 255) / 256;     // 7808

    // 3) t=0 cell (h0=c0=0)
    lstm_cell<<<cellBlocks, 256, 0, stream>>>(Xproj, Gh, cSt, hF, hB, 0);
    // 4) Gh = h1 @ W_hh^T ; t=1 cell
    wmma_gemm_bf16<<<gemmBlocks, 128, 0, stream>>>(hB, LSTM_HID, WhhB, LSTM_HID, nullptr,
                                                   Gh, GATES, waveCols, LSTM_HID);
    lstm_cell<<<cellBlocks, 256, 0, stream>>>(Xproj, Gh, cSt, hF, hB, 1);
    // 5) Gh = h2 @ W_hh^T ; t=2 cell -> h_last
    wmma_gemm_bf16<<<gemmBlocks, 128, 0, stream>>>(hB, LSTM_HID, WhhB, LSTM_HID, nullptr,
                                                   Gh, GATES, waveCols, LSTM_HID);
    lstm_cell<<<cellBlocks, 256, 0, stream>>>(Xproj, Gh, cSt, hF, hB, 2);

    // 6) a[s] = h_last . Wh ; c_all[r] = states . We
    row_dot<<<NSEQ, 256, 0, stream>>>(hF, LSTM_HID, fc_w, LSTM_HID, aV);
    row_dot<<<NSTATE, 256, 0, stream>>>(stF, ENC_DIM, fc_w + LSTM_HID, ENC_DIM, cAll);

    // 7) masked logsumexp loss + mean
    loss_kernel<<<NSEQ, 64, 0, stream>>>(aV, cAll, fc_b, loss);
    reduce_mean<<<1, 256, 0, stream>>>(loss, out);
}
